// Model_59777354826321
// MI455X (gfx1250) — matrix operand order repaired and checked by reading, not compiled, not run on hardware
//
#include <hip/hip_runtime.h>
#include <hip/hip_bf16.h>

typedef __attribute__((ext_vector_type(16))) _Float16 v16h;
typedef __attribute__((ext_vector_type(8)))  _Float16 v8h;
typedef __attribute__((ext_vector_type(8)))  float    v8f;

#define WMMA_F32_F16(a, b, c) \
    __builtin_amdgcn_wmma_f32_16x16x32_f16(false, (a), false, (b), (short)0, (c), false, false)

static constexpr int Bb   = 64;
static constexpr int Ss   = 1024;
static constexpr int Ee   = 32;
static constexpr int Hh   = 4;
static constexpr int Dd   = 8;
static constexpr int Vv   = 27;
static constexpr int LDSP = 40;   // padded LDS row stride (halves): conflict-free, 16B aligned rows
static constexpr int LK   = 40;   // padded K-block row stride (halves)

// ---------------------------------------------------------------------------
// Kernel 1: newlen[b] = min(prefix_length + 1, S)   (mask is a prefix mask)
// ---------------------------------------------------------------------------
__global__ void k_newlen(const float* __restrict__ mask, int* __restrict__ newlen) {
    __shared__ float red[256];
    const int b = blockIdx.x;
    float s = 0.f;
    for (int i = threadIdx.x; i < Ss; i += 256) s += mask[b * Ss + i];
    red[threadIdx.x] = s;
    __syncthreads();
    for (int st = 128; st > 0; st >>= 1) {
        if (threadIdx.x < st) red[threadIdx.x] += red[threadIdx.x + st];
        __syncthreads();
    }
    if (threadIdx.x == 0) {
        int len = (int)(red[0] + 0.5f);
        int nl  = len + 1;
        if (nl > Ss) nl = Ss;
        newlen[b] = nl;
    }
}

// ---------------------------------------------------------------------------
// Kernel 2: embedding + mask + packed QKV projection (WMMA, K = E = 32 exact).
// Writes Q (scaled by 1/sqrt(D)) as [B,H,S,D] f16, K transposed as [B,H,D,S]
// f16 (so attention B-matrix loads are contiguous), V as [B,H,S,D] f16.
// ---------------------------------------------------------------------------
__global__ __launch_bounds__(128) void k_qkv(
    const int*   __restrict__ x,
    const float* __restrict__ emb,
    const float* __restrict__ inw,   // [96,32] row-major
    const float* __restrict__ inb,   // [96]
    const int*   __restrict__ newlen,
    _Float16* __restrict__ Q,
    _Float16* __restrict__ Kt,
    _Float16* __restrict__ V) {
    const int lane = threadIdx.x & 31;
    const int wv   = threadIdx.x >> 5;
    const int tt   = blockIdx.x * 4 + wv;     // 16-token tile index
    const int t0   = tt * 16;
    const int b    = t0 >> 10;
    const int s0   = t0 & (Ss - 1);
    const int nl   = newlen[b];
    const int N    = lane & 15;
    const int hh   = lane >> 4;

    // B-matrices for in_proj: 6 tiles of 16 output features, col = lane%16, K = e in table order.
    v16h wB[6];
#pragma unroll
    for (int t = 0; t < 6; ++t) {
        v16h w;
#pragma unroll
        for (int n = 0; n < 16; ++n)
            w[n] = (_Float16)inw[(t * 16 + N) * Ee + ((n < 8) ? (hh * 8 + n) : (16 + hh * 8 + n - 8))];
        wB[t] = w;
    }

    // A-matrix: h = emb[x] * newmask, f16, 16 tokens x 32 dims.
    const int tok = t0 + N;
    const int idx = x[tok];
    const float vm = ((s0 + N) < nl) ? 1.f : 0.f;
    const int e0 = hh * 8;        // lanes 0-15: K 0..7 ; lanes 16-31: K 8..15
    const int e1 = 16 + hh * 8;   // lanes 0-15: K16..23; lanes 16-31: K24..31
    v16h ah;
#pragma unroll
    for (int i = 0; i < 8; ++i) {
        ah[i]     = (_Float16)(emb[idx * Ee + e0 + i] * vm);
        ah[8 + i] = (_Float16)(emb[idx * Ee + e1 + i] * vm);
    }

    const float qscale = 0.35355339059327373f;  // 1/sqrt(8)
#pragma unroll
    for (int t = 0; t < 6; ++t) {
        const float bf = inb[t * 16 + N];
        v8f c;
#pragma unroll
        for (int r = 0; r < 8; ++r) c[r] = bf;
        c = WMMA_F32_F16(ah, wB[t], c);   // C layout: row M=r+8*hh, col f=t*16+N

        if (t < 2) {                      // Q features 0..31
            const int f = t * 16 + N, head = f >> 3, d = f & 7;
#pragma unroll
            for (int r = 0; r < 8; ++r) {
                const int s = s0 + r + 8 * hh;
                Q[((b * Hh + head) * Ss + s) * Dd + d] = (_Float16)(c[r] * qscale);
            }
        } else if (t < 4) {               // K features 32..63 -> transposed store
            const int e = (t - 2) * 16 + N, head = e >> 3, d = e & 7;
#pragma unroll
            for (int r = 0; r < 8; ++r) {
                const int s = s0 + r + 8 * hh;
                Kt[((b * Hh + head) * Dd + d) * Ss + s] = (_Float16)c[r];
            }
        } else {                          // V features 64..95
            const int e = (t - 4) * 16 + N, head = e >> 3, d = e & 7;
#pragma unroll
            for (int r = 0; r < 8; ++r) {
                const int s = s0 + r + 8 * hh;
                V[((b * Hh + head) * Ss + s) * Dd + d] = (_Float16)c[r];
            }
        }
    }
}

// ---------------------------------------------------------------------------
// Kernel 3: flash attention. One wave per (b,h,16-query tile), 4 waves/block.
// K/V blocks are double-buffered into LDS with global_load_async_to_lds_b128
// (ASYNCcnt pipelining: block j+1 copies overlap block j compute).
// The V B-matrix carries a ones-column at N=8, so the P·V WMMA also produces
// the softmax denominator in accumulator column 8 (no sum shuffles needed).
// ---------------------------------------------------------------------------
__global__ __launch_bounds__(128) void k_attn(
    const _Float16* __restrict__ Q,
    const _Float16* __restrict__ Kt,
    const _Float16* __restrict__ V,
    const int* __restrict__ newlen,
    _Float16* __restrict__ ctx) {     // [B,S,E] f16
    __shared__ __align__(16) _Float16 lds_p[4][16 * LDSP];
    __shared__ __align__(16) _Float16 lds_k[4][2][8 * LK];    // 8 dims x 32 keys, padded
    __shared__ __align__(16) _Float16 lds_v[4][2][32 * Dd];   // 32 keys x 8 dims

    const int lane  = threadIdx.x & 31;
    const int wv    = threadIdx.x >> 5;
    const int blk   = blockIdx.x;
    const int b     = blk >> 6;
    const int h     = (blk >> 4) & 3;
    const int chunk = blk & 15;
    const int q0    = chunk * 64 + wv * 16;
    const int nl    = newlen[b];
    const int N     = lane & 15;
    const int hh    = lane >> 4;

    const _Float16* KtB = Kt + (b * Hh + h) * Dd * Ss;
    const _Float16* VB  = V  + (b * Hh + h) * Ss * Dd;

    // Async copy of one 32-key K/V block into LDS buffer `buf`.
    // K: 8 rows x 64B -> 32 lanes x 16B.  V: 32 rows x 16B -> 32 lanes x 16B.
    auto issue_async = [&](int jj, int buf) {
        const int k0n = jj * 32;
        unsigned koff = (unsigned)(uintptr_t)(&lds_k[wv][buf][(lane >> 2) * LK + (lane & 3) * 8]);
        const _Float16* kg = KtB + (lane >> 2) * Ss + k0n + (lane & 3) * 8;
        asm volatile("global_load_async_to_lds_b128 %0, %1, off"
                     :: "v"(koff), "v"(kg) : "memory");
        unsigned voff = (unsigned)(uintptr_t)(&lds_v[wv][buf][lane * Dd]);
        const _Float16* vg = VB + (k0n + lane) * Dd;
        asm volatile("global_load_async_to_lds_b128 %0, %1, off"
                     :: "v"(voff), "v"(vg) : "memory");
    };

    // A-matrix: Q tile, head-dim 8 zero-padded to K=32.
    v16h aq = {};
    if (lane < 16) {
        const v8h q8 = *(const v8h*)(Q + ((b * Hh + h) * Ss + q0 + lane) * Dd);
#pragma unroll
        for (int i = 0; i < 8; ++i) aq[i] = q8[i];
    }

    v8f acc = {};                 // cols 0..7 = ctx, col 8 = running denominator
    float m[8];
#pragma unroll
    for (int r = 0; r < 8; ++r) m[r] = -1e30f;

    const int jmax = (nl + 31) >> 5;
    issue_async(0, 0);

    for (int j = 0; j < jmax; ++j) {
        const int buf = j & 1;
        const int k0  = j * 32;
        if (j + 1 < jmax) {
            issue_async(j + 1, buf ^ 1);
            asm volatile("s_wait_asynccnt 0x2" ::: "memory");   // oldest batch done
        } else {
            asm volatile("s_wait_asynccnt 0x0" ::: "memory");
        }

        // B-matrices from LDS: lane = output column, elements = contraction index (table order).
        v16h bk0 = {}, bk1 = {};
        if (lane < 16) {
            const _Float16* kr = &lds_k[wv][buf][0];
#pragma unroll
            for (int i = 0; i < 8; ++i) {
                bk0[i] = kr[i * LK + N];
                bk1[i] = kr[i * LK + 16 + N];
            }
        }
        v16h bv = {};
        {
#pragma unroll
            for (int s = 0; s < 16; ++s) {
                const int kk = (s < 8) ? (hh * 8 + s) : (16 + hh * 8 + s - 8);
                bv[s] = (N < 8) ? lds_v[wv][buf][kk * Dd + N] : ((N == 8) ? (_Float16)1.0f : (_Float16)0.f);   // ones-column -> row sums in acc col 8
            }
        }

        v8f zc = {};
        v8f s0v = WMMA_F32_F16(aq, bk0, zc);
        v8f s1v = WMMA_F32_F16(aq, bk1, zc);

        const bool ms0 = (k0 + N) >= nl;
        const bool ms1 = (k0 + 16 + N) >= nl;
        float p0[8], p1[8];
#pragma unroll
        for (int r = 0; r < 8; ++r) {
            if (ms0) s0v[r] = -1e30f;
            if (ms1) s1v[r] = -1e30f;
        }
        // online softmax: row max across 16-lane halves
#pragma unroll
        for (int r = 0; r < 8; ++r) {
            float t = fmaxf(s0v[r], s1v[r]);
            t = fmaxf(t, __shfl_xor(t, 1, 32));
            t = fmaxf(t, __shfl_xor(t, 2, 32));
            t = fmaxf(t, __shfl_xor(t, 4, 32));
            t = fmaxf(t, __shfl_xor(t, 8, 32));
            const float mn  = fmaxf(m[r], t);
            const float fac = __expf(m[r] - mn);
            m[r] = mn;
            p0[r] = __expf(s0v[r] - mn);
            p1[r] = __expf(s1v[r] - mn);
            acc[r] = acc[r] * fac;
        }

        // C-layout P -> A-layout P(16x32 f16) via LDS
        _Float16* lp = &lds_p[wv][0];
#pragma unroll
        for (int r = 0; r < 8; ++r) {
            const int row = r + 8 * hh;
            lp[row * LDSP + N]      = (_Float16)p0[r];
            lp[row * LDSP + 16 + N] = (_Float16)p1[r];
        }
        asm volatile("s_wait_dscnt 0" ::: "memory");
        v16h ap;
        {
            const v8h lo = *(const v8h*)(lp + N * LDSP + hh * 8);
            const v8h hi = *(const v8h*)(lp + N * LDSP + 16 + hh * 8);
#pragma unroll
            for (int i = 0; i < 8; ++i) { ap[i] = lo[i]; ap[8 + i] = hi[i]; }
        }

        acc = WMMA_F32_F16(ap, bv, acc);
    }

    // denominator lives in column N==8 of each half -> broadcast, divide, store
#pragma unroll
    for (int r = 0; r < 8; ++r) {
        const float l_r = __shfl(acc[r], hh * 16 + 8, 32);
        if (N < 8) {
            const int q = q0 + r + 8 * hh;
            ctx[(b * Ss + q) * Ee + h * Dd + N] = (_Float16)(acc[r] / l_r);
        }
    }
}

// ---------------------------------------------------------------------------
// Kernel 4: out_proj (32x32) + fc (27x32), both K=32 WMMA; intermediate
// re-shaped C->A through LDS. One wave per 16-token tile.
// ---------------------------------------------------------------------------
__global__ __launch_bounds__(128) void k_proj(
    const _Float16* __restrict__ ctx,
    const float* __restrict__ wout,  // [32,32]
    const float* __restrict__ bout,  // [32]
    const float* __restrict__ wfc,   // [27,32]
    const float* __restrict__ bfc,   // [27]
    float* __restrict__ out) {       // [B,S,27]
    __shared__ __align__(16) _Float16 lds_t[4][16 * LDSP];

    const int lane = threadIdx.x & 31;
    const int wv   = threadIdx.x >> 5;
    const int t0   = (blockIdx.x * 4 + wv) * 16;
    const int N    = lane & 15;
    const int hh   = lane >> 4;

    v16h w0, w1, f0, f1;
#pragma unroll
    for (int n = 0; n < 16; ++n) {
        const int kk = (n < 8) ? (hh * 8 + n) : (16 + hh * 8 + n - 8);
        w0[n] = (_Float16)wout[N * Ee + kk];
        w1[n] = (_Float16)wout[(16 + N) * Ee + kk];
        f0[n] = (_Float16)wfc[N * Ee + kk];
        f1[n] = (N < Vv - 16) ? (_Float16)wfc[(16 + N) * Ee + kk] : (_Float16)0.f;
    }

    v16h ac;
    {
        const int tok = t0 + N;
        const v8h lo = *(const v8h*)(ctx + tok * Ee + hh * 8);
        const v8h hi = *(const v8h*)(ctx + tok * Ee + 16 + hh * 8);
#pragma unroll
        for (int i = 0; i < 8; ++i) { ac[i] = lo[i]; ac[8 + i] = hi[i]; }
    }

    v8f c0, c1;
    {
        const float b0 = bout[N], b1 = bout[16 + N];
#pragma unroll
        for (int r = 0; r < 8; ++r) { c0[r] = b0; c1[r] = b1; }
    }
    v8f t0v = WMMA_F32_F16(ac, w0, c0);
    v8f t1v = WMMA_F32_F16(ac, w1, c1);

    _Float16* lt = &lds_t[wv][0];
#pragma unroll
    for (int r = 0; r < 8; ++r) {
        const int row = r + 8 * hh;
        lt[row * LDSP + N]      = (_Float16)t0v[r];
        lt[row * LDSP + 16 + N] = (_Float16)t1v[r];
    }
    asm volatile("s_wait_dscnt 0" ::: "memory");
    v16h at;
    {
        const v8h lo = *(const v8h*)(lt + N * LDSP + hh * 8);
        const v8h hi = *(const v8h*)(lt + N * LDSP + 16 + hh * 8);
#pragma unroll
        for (int i = 0; i < 8; ++i) { at[i] = lo[i]; at[8 + i] = hi[i]; }
    }

    v8f cf0, cf1;
    {
        const float b0 = bfc[N];
        const float b1 = (N < Vv - 16) ? bfc[16 + N] : 0.f;
#pragma unroll
        for (int r = 0; r < 8; ++r) { cf0[r] = b0; cf1[r] = b1; }
    }
    v8f lo0 = WMMA_F32_F16(at, f0, cf0);
    v8f lo1 = WMMA_F32_F16(at, f1, cf1);

#pragma unroll
    for (int r = 0; r < 8; ++r) {
        const int tok = t0 + r + 8 * hh;
        out[tok * Vv + N] = lo0[r];
        if (N < Vv - 16) out[tok * Vv + 16 + N] = lo1[r];
    }
}

// ---------------------------------------------------------------------------
extern "C" void kernel_launch(void* const* d_in, const int* in_sizes, int n_in,
                              void* d_out, int out_size, void* d_ws, size_t ws_size,
                              hipStream_t stream) {
    const int*   x    = (const int*)  d_in[0];
    const float* mask = (const float*)d_in[1];
    const float* emb  = (const float*)d_in[2];
    const float* inw  = (const float*)d_in[3];
    const float* inb  = (const float*)d_in[4];
    const float* outw = (const float*)d_in[5];
    const float* outb = (const float*)d_in[6];
    const float* fcw  = (const float*)d_in[7];
    const float* fcb  = (const float*)d_in[8];

    char* ws = (char*)d_ws;
    int*       newlen = (int*)ws;
    _Float16*  Q      = (_Float16*)(ws + 256);
    const size_t QS   = (size_t)Bb * Hh * Ss * Dd;   // 2M halves = 4MB each
    _Float16*  Kt     = Q + QS;
    _Float16*  V      = Kt + QS;
    _Float16*  CTX    = V + QS;                       // [B,S,E] f16

    k_newlen<<<Bb, 256, 0, stream>>>(mask, newlen);
    k_qkv   <<<(Bb * Ss / 16) / 4, 128, 0, stream>>>(x, emb, inw, inb, newlen, Q, Kt, V);
    k_attn  <<<Bb * Hh * (Ss / 64), 128, 0, stream>>>(Q, Kt, V, newlen, CTX);
    k_proj  <<<(Bb * Ss / 16) / 4, 128, 0, stream>>>(CTX, outw, outb, fcw, fcb, (float*)d_out);
}
